// SparseLogicBlock_28346784153815
// MI455X (gfx1250) — 8-bit matrix operand order repaired and checked by reading, compiled, not run on hardware
//
#include <hip/hip_runtime.h>
#include <hip/hip_bf16.h>
#include <cstdint>
#include <cstddef>


typedef int v8i __attribute__((ext_vector_type(8)));

constexpr int Bn = 16, Ln = 256, Dn = 1024, DHn = 4096;
constexpr int Mn = Bn * Ln;
constexpr int MIXn = 25 * Dn;

__constant__ int c_off[25] = {0, 2, 13, 22, 6, 11, 25, 7, 18, 3, 17, 19, 10,
                              0, 32, 64, 96, 128, 160, 192, 224, 1, 2, 3, 4};
__constant__ int c_rot[25] = {1, 1, 1, 1, 1, 1, 1, 1, 1, 0, 1, 1, 0,
                              1, 1, 1, 1, 1, 1, 1, 1, 0, 0, 0, 0};

constexpr size_t OFF_SCALES = 0;
constexpr size_t OFF_PART   = 4096;
constexpr size_t OFF_S1     = 0x08000;
constexpr size_t OFF_S2     = 0x0C000;
constexpr size_t OFF_S3     = 0x10000;
constexpr size_t OFF_RMAX   = 0x14000;
constexpr size_t OFF_XN     = 0x20000;
constexpr size_t OFF_XNEW   = OFF_XN   + (size_t)Mn * Dn * 4;
constexpr size_t OFF_WQG    = OFF_XNEW + (size_t)Mn * Dn * 4;
constexpr size_t OFF_WQ1    = OFF_WQG  + (size_t)Dn * MIXn;
constexpr size_t OFF_WQ2    = OFF_WQ1  + (size_t)DHn * Dn;
constexpr size_t OFF_AG     = OFF_WQ2  + (size_t)Dn * DHn;
constexpr size_t OFF_A1     = OFF_AG   + (size_t)Mn * MIXn;
constexpr size_t OFF_A2     = OFF_A1   + (size_t)Mn * Dn;
constexpr size_t OFF_H      = OFF_A2   + (size_t)Mn * DHn;
constexpr size_t OFF_HC     = OFF_H    + (size_t)Mn * DHn * 4;

__device__ inline float block_reduce_sum(float v, float* red) {
  int t = threadIdx.x;
  red[t] = v; __syncthreads();
  for (int s = 128; s > 0; s >>= 1) {
    if (t < s) red[t] += red[t + s];
    __syncthreads();
  }
  float r = red[0]; __syncthreads();
  return r;
}

__device__ inline float block_reduce_max(float v, float* red) {
  int t = threadIdx.x;
  red[t] = v; __syncthreads();
  for (int s = 128; s > 0; s >>= 1) {
    if (t < s) red[t] = fmaxf(red[t], red[t + s]);
    __syncthreads();
  }
  float r = red[0]; __syncthreads();
  return r;
}

__device__ inline v8i load_frag(const char* p) {
  const int4* q = (const int4*)p;
  int4 a = q[0], b = q[1];
  v8i r;
  r[0] = a.x; r[1] = a.y; r[2] = a.z; r[3] = a.w;
  r[4] = b.x; r[5] = b.y; r[6] = b.z; r[7] = b.w;
  return r;
}

__global__ __launch_bounds__(256) void abssum_partial_kernel(
    const float* __restrict__ w, float* __restrict__ part, size_t count) {
  __shared__ float red[256];
  float s = 0.f;
  for (size_t i = (size_t)blockIdx.x * 256 + threadIdx.x; i < count;
       i += (size_t)gridDim.x * 256)
    s += fabsf(w[i]);
  float r = block_reduce_sum(s, red);
  if (threadIdx.x == 0) part[blockIdx.x] = r;
}

__global__ __launch_bounds__(256) void absmean_final_kernel(
    const float* __restrict__ part, int nblk, float count,
    float* __restrict__ ws_out) {
  __shared__ float red[256];
  float s = 0.f;
  for (int i = threadIdx.x; i < nblk; i += 256) s += part[i];
  float r = block_reduce_sum(s, red);
  if (threadIdx.x == 0) {
    float mean = r / count;
    ws_out[0] = 1.0f / fmaxf(mean, 1e-5f);
  }
}

__global__ __launch_bounds__(256) void quantB_kernel(
    const float* __restrict__ W, const float* __restrict__ wsp,
    unsigned int* __restrict__ Bf, int N, int K) {
  size_t gid = (size_t)blockIdx.x * 256 + threadIdx.x;
  size_t total = (size_t)N * K / 4;
  if (gid >= total) return;
  int nkt = K >> 6;
  size_t tileIdx = gid >> 8;
  int r = (int)(gid & 255);
  int lane = r >> 3, v = r & 7;
  int ntile = (int)(tileIdx / nkt), ktile = (int)(tileIdx % nkt);
  int n = ntile * 16 + (lane & 15);
  int half = lane >> 4;
  int kb = ktile * 64 + ((v >> 2) << 5) + (half << 4) + ((v & 3) << 2);
  float wsv = wsp[0];
  unsigned int packed = 0;
  for (int j = 0; j < 4; ++j) {
    float w = W[(size_t)n * K + kb + j];
    float q = rintf(w * wsv);
    q = fminf(fmaxf(q, -1.0f), 1.0f);
    packed |= ((unsigned int)((int)q & 0xff)) << (8 * j);
  }
  Bf[gid] = packed;
}

__global__ __launch_bounds__(256) void ln_kernel(
    const float* __restrict__ x, const float* __restrict__ g,
    const float* __restrict__ b, float* __restrict__ xn,
    float* __restrict__ rowmax) {
  __shared__ float red[256];
  int row = blockIdx.x;
  const float* xr = x + (size_t)row * Dn;
  float v[4], s = 0.f, s2 = 0.f;
#pragma unroll
  for (int i = 0; i < 4; ++i) {
    v[i] = xr[threadIdx.x + i * 256];
    s += v[i];
    s2 += v[i] * v[i];
  }
  float sum = block_reduce_sum(s, red);
  float sq  = block_reduce_sum(s2, red);
  float mu  = sum * (1.0f / Dn);
  float var = sq * (1.0f / Dn) - mu * mu;
  float inv = rsqrtf(var + 1e-5f);
  float mx = 0.f;
#pragma unroll
  for (int i = 0; i < 4; ++i) {
    int d = threadIdx.x + i * 256;
    float y = (v[i] - mu) * inv * g[d] + b[d];
    xn[(size_t)row * Dn + d] = y;
    mx = fmaxf(mx, fabsf(y));
  }
  float m = block_reduce_max(mx, red);
  if (threadIdx.x == 0) rowmax[row] = m;
}

__global__ __launch_bounds__(256) void scale_kernel(
    const float* __restrict__ rowmax, float* __restrict__ s, int gather) {
  int m = blockIdx.x * 256 + threadIdx.x;
  if (m >= Mn) return;
  float mx;
  if (gather) {
    int b = m >> 8, l = m & 255;
    mx = 0.f;
    for (int g = 0; g < 25; ++g) {
      int off = c_off[g];
      int src; bool ok = true;
      if (c_rot[g]) src = (l + off) & 255;
      else { src = l + off; if (src >= Ln) { ok = false; src = 0; } }
      if (ok) mx = fmaxf(mx, rowmax[(b << 8) + src]);
    }
  } else {
    mx = rowmax[m];
  }
  s[m] = 127.0f / fmaxf(mx, 1e-5f);
}

__global__ __launch_bounds__(256) void quantA_kernel(
    const float* __restrict__ src, const float* __restrict__ s,
    unsigned int* __restrict__ Af, int K, int gather) {
  size_t gid = (size_t)blockIdx.x * 256 + threadIdx.x;
  size_t total = (size_t)Mn * K / 4;
  if (gid >= total) return;
  int nkt = K >> 6;
  size_t tileIdx = gid >> 8;
  int r = (int)(gid & 255);
  int lane = r >> 3, v = r & 7;
  int mtile = (int)(tileIdx / nkt), ktile = (int)(tileIdx % nkt);
  int mrow = mtile * 16 + (lane & 15);
  int half = lane >> 4;
  int kb = ktile * 64 + ((v >> 2) << 5) + (half << 4) + ((v & 3) << 2);
  float sv = s[mrow];
  int b = mrow >> 8, l = mrow & 255;
  unsigned int packed = 0;
  for (int j = 0; j < 4; ++j) {
    int k = kb + j;
    float val;
    if (gather) {
      int g = k >> 10, d = k & 1023;
      int off = c_off[g];
      int srcl; bool ok = true;
      if (c_rot[g]) srcl = (l + off) & 255;
      else { srcl = l + off; if (srcl >= Ln) { ok = false; srcl = 0; } }
      val = ok ? src[((size_t)((b << 8) + srcl)) * Dn + d] : 0.0f;
    } else {
      val = src[(size_t)mrow * K + k];
    }
    float q = rintf(val * sv);
    q = fminf(fmaxf(q, -128.0f), 127.0f);
    packed |= ((unsigned int)((int)q & 0xff)) << (8 * j);
  }
  Af[gid] = packed;
}

__global__ __launch_bounds__(256) void gemm_i8_kernel(
    const char* __restrict__ Af, const char* __restrict__ Bf,
    float* __restrict__ out, const float* __restrict__ srow,
    const float* __restrict__ wsp, const float* __restrict__ resid,
    int N, int nkt) {
  int tid = threadIdx.x;
  int lane = tid & 31, wave = tid >> 5;
  int waveM = wave & 1, waveN = wave >> 1;
  int mtile0 = blockIdx.y * 8 + waveM * 4;
  int ntile0 = blockIdx.x * 16 + waveN * 4;

  unsigned aoff[4], boff[4];
#pragma unroll
  for (int i = 0; i < 4; ++i)
    aoff[i] = ((unsigned)(mtile0 + i) * (unsigned)nkt) * 1024u +
              (unsigned)lane * 32u;
#pragma unroll
  for (int j = 0; j < 4; ++j)
    boff[j] = ((unsigned)(ntile0 + j) * (unsigned)nkt) * 1024u +
              (unsigned)lane * 32u;

  v8i acc[4][4];
#pragma unroll
  for (int i = 0; i < 4; ++i)
#pragma unroll
    for (int j = 0; j < 4; ++j) acc[i][j] = v8i{0, 0, 0, 0, 0, 0, 0, 0};

#pragma unroll 2
  for (int kt = 0; kt < nkt; ++kt) {
    unsigned ko = (unsigned)kt * 1024u;
    if (kt + 1 < nkt) {
      __builtin_prefetch(Af + (aoff[0] + ko + 1024u), 0, 1);
      __builtin_prefetch(Bf + (boff[0] + ko + 1024u), 0, 1);
    }
    v8i a[4], b[4];
#pragma unroll
    for (int i = 0; i < 4; ++i) a[i] = load_frag(Af + (aoff[i] + ko));
#pragma unroll
    for (int j = 0; j < 4; ++j) b[j] = load_frag(Bf + (boff[j] + ko));
#pragma unroll
    for (int i = 0; i < 4; ++i)
#pragma unroll
      for (int j = 0; j < 4; ++j)
        acc[i][j] = __builtin_amdgcn_wmma_i32_16x16x64_iu8(
            true, a[i], true, b[j], acc[i][j], false, false);
  }

  float wsv = wsp[0];
  int half = lane >> 4, nIn = lane & 15;
#pragma unroll
  for (int i = 0; i < 4; ++i) {
    int rowBase = (mtile0 + i) * 16 + half * 8;
#pragma unroll
    for (int r = 0; r < 8; ++r) {
      int row = rowBase + r;
      float f = 1.0f / (srow[row] * wsv);
#pragma unroll
      for (int j = 0; j < 4; ++j) {
        int col = (ntile0 + j) * 16 + nIn;
        size_t o = (size_t)row * N + col;
        float v = (float)acc[i][j][r] * f;
        out[o] = (resid ? resid[o] : 0.0f) + v;
      }
    }
  }
}

__global__ __launch_bounds__(256) void conv_silu_kernel(
    const float* __restrict__ h, const float* __restrict__ cw,
    const float* __restrict__ cb, float* __restrict__ hc,
    float* __restrict__ rowmax) {
  __shared__ float red[256];
  int m = blockIdx.x;
  int l = m & 255;
  const float* h0 = h + (size_t)m * DHn;
  bool hasP = (l > 0), hasN = (l < Ln - 1);
  float mx = 0.f;
  for (int c = threadIdx.x; c < DHn; c += 256) {
    float y = cw[c * 3 + 1] * h0[c] + cb[c];
    if (hasP) y += cw[c * 3 + 0] * h0[c - DHn];
    if (hasN) y += cw[c * 3 + 2] * h0[c + DHn];
    float z = y / (1.0f + __expf(-y));
    hc[(size_t)m * DHn + c] = z;
    mx = fmaxf(mx, fabsf(z));
  }
  float r = block_reduce_max(mx, red);
  if (threadIdx.x == 0) rowmax[m] = r;
}

extern "C" void kernel_launch(void* const* d_in, const int* in_sizes, int n_in,
                              void* d_out, int out_size, void* d_ws,
                              size_t ws_size, hipStream_t stream) {
  const float* x      = (const float*)d_in[0];
  const float* ln1_g  = (const float*)d_in[1];
  const float* ln1_b  = (const float*)d_in[2];
  const float* ln2_g  = (const float*)d_in[3];
  const float* ln2_b  = (const float*)d_in[4];
  const float* w_gate = (const float*)d_in[5];
  const float* w1     = (const float*)d_in[6];
  const float* conv_w = (const float*)d_in[7];
  const float* conv_b = (const float*)d_in[8];
  const float* w2     = (const float*)d_in[9];

  char* ws = (char*)d_ws;
  float* scales = (float*)(ws + OFF_SCALES);
  float* part   = (float*)(ws + OFF_PART);
  float* s1     = (float*)(ws + OFF_S1);
  float* s2     = (float*)(ws + OFF_S2);
  float* s3     = (float*)(ws + OFF_S3);
  float* rmax   = (float*)(ws + OFF_RMAX);
  float* xn     = (float*)(ws + OFF_XN);
  float* xnew   = (float*)(ws + OFF_XNEW);
  char*  wqg    = ws + OFF_WQG;
  char*  wq1    = ws + OFF_WQ1;
  char*  wq2    = ws + OFF_WQ2;
  char*  Ag     = ws + OFF_AG;
  char*  A1     = ws + OFF_A1;
  char*  A2     = ws + OFF_A2;
  float* h      = (float*)(ws + OFF_H);
  float* hc     = (float*)(ws + OFF_HC);

  size_t cg = (size_t)Dn * MIXn, c1 = (size_t)DHn * Dn, c2 = (size_t)Dn * DHn;
  abssum_partial_kernel<<<1024, 256, 0, stream>>>(w_gate, part, cg);
  absmean_final_kernel<<<1, 256, 0, stream>>>(part, 1024, (float)cg, scales + 0);
  abssum_partial_kernel<<<1024, 256, 0, stream>>>(w1, part, c1);
  absmean_final_kernel<<<1, 256, 0, stream>>>(part, 1024, (float)c1, scales + 1);
  abssum_partial_kernel<<<1024, 256, 0, stream>>>(w2, part, c2);
  absmean_final_kernel<<<1, 256, 0, stream>>>(part, 1024, (float)c2, scales + 2);

  quantB_kernel<<<(unsigned)(cg / 4 / 256), 256, 0, stream>>>(
      w_gate, scales + 0, (unsigned int*)wqg, Dn, MIXn);
  quantB_kernel<<<(unsigned)(c1 / 4 / 256), 256, 0, stream>>>(
      w1, scales + 1, (unsigned int*)wq1, DHn, Dn);
  quantB_kernel<<<(unsigned)(c2 / 4 / 256), 256, 0, stream>>>(
      w2, scales + 2, (unsigned int*)wq2, Dn, DHn);

  ln_kernel<<<Mn, 256, 0, stream>>>(x, ln1_g, ln1_b, xn, rmax);
  scale_kernel<<<Mn / 256, 256, 0, stream>>>(rmax, s1, 1);
  quantA_kernel<<<(unsigned)((size_t)Mn * MIXn / 4 / 256), 256, 0, stream>>>(
      xn, s1, (unsigned int*)Ag, MIXn, 1);
  gemm_i8_kernel<<<dim3(Dn / 256, Mn / 128), 256, 0, stream>>>(
      Ag, wqg, xnew, s1, scales + 0, x, Dn, MIXn / 64);

  ln_kernel<<<Mn, 256, 0, stream>>>(xnew, ln2_g, ln2_b, xn, rmax);
  scale_kernel<<<Mn / 256, 256, 0, stream>>>(rmax, s2, 0);
  quantA_kernel<<<(unsigned)((size_t)Mn * Dn / 4 / 256), 256, 0, stream>>>(
      xn, s2, (unsigned int*)A1, Dn, 0);
  gemm_i8_kernel<<<dim3(DHn / 256, Mn / 128), 256, 0, stream>>>(
      A1, wq1, h, s2, scales + 1, nullptr, DHn, Dn / 64);

  conv_silu_kernel<<<Mn, 256, 0, stream>>>(h, conv_w, conv_b, hc, rmax);
  scale_kernel<<<Mn / 256, 256, 0, stream>>>(rmax, s3, 0);
  quantA_kernel<<<(unsigned)((size_t)Mn * DHn / 4 / 256), 256, 0, stream>>>(
      hc, s3, (unsigned int*)A2, DHn, 0);
  gemm_i8_kernel<<<dim3(Dn / 256, Mn / 128), 256, 0, stream>>>(
      A2, wq2, (float*)d_out, s3, scales + 2, xnew, Dn, DHn / 64);
}
